// TransformerEncoder_20091857011463
// MI455X (gfx1250) — compile-verified, matrix operand order checked by reading, not run on hardware
//
#include <hip/hip_runtime.h>
#include <hip/hip_bf16.h>

// ---------------------------------------------------------------------------
// CDNA5 (gfx1250) types for WMMA
// ---------------------------------------------------------------------------
typedef __attribute__((ext_vector_type(16))) __bf16 v16bf;
typedef __attribute__((ext_vector_type(8)))  float  v8f;

union FragBF {
    v16bf v;
    uint4 q[2];
};

static __device__ inline v8f vzero8() {
    v8f z = {0.f, 0.f, 0.f, 0.f, 0.f, 0.f, 0.f, 0.f};
    return z;
}

// fp32 -> bf16 round-to-nearest-even (stored as ushort)
static __device__ inline unsigned short f2bf(float f) {
    unsigned int u = __float_as_uint(f);
    u += 0x7FFFu + ((u >> 16) & 1u);
    return (unsigned short)(u >> 16);
}
static __device__ inline float bf2f(unsigned short h) {
    return __uint_as_float(((unsigned int)h) << 16);
}

static __device__ inline float gelu_exact(float x) {
    return 0.5f * x * (1.0f + erff(x * 0.70710678118654752f));
}
static __device__ inline float gelu_tanh(float x) {
    return 0.5f * x * (1.0f + tanhf(0.7978845608028654f * (x + 0.044715f * x * x * x)));
}

// fixed sinusoidal position embedding, D=1024
static __device__ inline float posemb(int s, int n) {
    const float LOG1E4 = 9.210340371976184f;
    if (n < 512) return __sinf((float)s * __expf(-LOG1E4 * (float)n * (1.0f / 512.0f)));
    return __cosf((float)s * __expf(-LOG1E4 * (float)(n - 512) * (1.0f / 512.0f)));
}

// ---------------------------------------------------------------------------
// Time-embedding MLP (B=8 rows -> negligible cost, scalar kernels)
// ---------------------------------------------------------------------------
__global__ __launch_bounds__(256)
void time_mlp1(const float* __restrict__ timev, const float* __restrict__ W1,
               const float* __restrict__ b1, float* __restrict__ hid) {
    __shared__ float t0[1024];
    const int tid = threadIdx.x, b = blockIdx.y;
    const float t = timev[b];
#pragma unroll
    for (int i = 0; i < 4; ++i) {
        int d = tid + i * 256;
        int k = (d < 512) ? d : d - 512;
        float inv = __expf(-9.210340371976184f / 511.0f * (float)k);
        float ph  = t * inv;
        t0[d] = (d < 512) ? __sinf(ph) : __cosf(ph);
    }
    __syncthreads();
    int n = blockIdx.x * 256 + tid;
    float acc = b1[n];
    for (int d = 0; d < 1024; ++d) acc = fmaf(t0[d], W1[(size_t)d * 4096 + n], acc);
    hid[(size_t)b * 4096 + n] = gelu_exact(acc);
}

__global__ __launch_bounds__(256)
void time_mlp2(const float* __restrict__ hid, const float* __restrict__ W2,
               const float* __restrict__ b2, float* __restrict__ temb) {
    __shared__ float t0[4096];
    const int tid = threadIdx.x, b = blockIdx.y;
#pragma unroll
    for (int i = 0; i < 16; ++i) t0[tid + i * 256] = hid[(size_t)b * 4096 + tid + i * 256];
    __syncthreads();
    int n = blockIdx.x * 256 + tid;
    float acc = b2[n];
    for (int d = 0; d < 4096; ++d) acc = fmaf(t0[d], W2[(size_t)d * 1024 + n], acc);
    temb[(size_t)b * 1024 + n] = gelu_exact(acc);
}

// ---------------------------------------------------------------------------
// fp32 -> bf16 conversion
// ---------------------------------------------------------------------------
__global__ __launch_bounds__(256)
void cvt_bf16(const float* __restrict__ in, unsigned short* __restrict__ o, int n) {
    int i = blockIdx.x * 256 + threadIdx.x;
    if (i < n) o[i] = f2bf(in[i]);
}

// ---------------------------------------------------------------------------
// LayerNorm (D=1024) -> bf16
// ---------------------------------------------------------------------------
__global__ __launch_bounds__(256)
void layernorm_bf16(const float* __restrict__ h, const float* __restrict__ g,
                    const float* __restrict__ bt, unsigned short* __restrict__ u) {
    const int row = blockIdx.x, tid = threadIdx.x;
    const float* x = h + (size_t)row * 1024;
    float v[4];
    float s = 0.f, sq = 0.f;
#pragma unroll
    for (int i = 0; i < 4; ++i) {
        v[i] = x[tid + i * 256];
        s += v[i];
        sq += v[i] * v[i];
    }
#pragma unroll
    for (int off = 16; off >= 1; off >>= 1) {
        s  += __shfl_xor(s, off, 32);
        sq += __shfl_xor(sq, off, 32);
    }
    __shared__ float rs[8], rq[8];
    if ((tid & 31) == 0) { rs[tid >> 5] = s; rq[tid >> 5] = sq; }
    __syncthreads();
    s = 0.f; sq = 0.f;
#pragma unroll
    for (int i = 0; i < 8; ++i) { s += rs[i]; sq += rq[i]; }
    float mean = s * (1.f / 1024.f);
    float var  = sq * (1.f / 1024.f) - mean * mean;
    float rstd = rsqrtf(var + 1e-5f);
#pragma unroll
    for (int i = 0; i < 4; ++i) {
        int c = tid + i * 256;
        u[(size_t)row * 1024 + c] = f2bf((v[i] - mean) * rstd * g[c] + bt[c]);
    }
}

// ---------------------------------------------------------------------------
// ffg = ff * gelu_tanh(gate), read from proj (bf16, pitch 11264)
// ---------------------------------------------------------------------------
__global__ __launch_bounds__(256)
void ffg_kernel(const unsigned short* __restrict__ proj, unsigned short* __restrict__ outp) {
    size_t i   = (size_t)blockIdx.x * 256 + threadIdx.x;
    size_t row = i >> 12;
    int    col = (int)(i & 4095);
    size_t base = row * 11264;
    float ff = bf2f(proj[base + 3072 + col]);
    float gt = bf2f(proj[base + 7168 + col]);
    outp[i] = f2bf(ff * gelu_tanh(gt));
}

// ---------------------------------------------------------------------------
// Tiled bf16 WMMA GEMM: C[M,N] = A[M,K](bf16) * B[K,N](fp32->bf16) + epilogue
// Block tile 128x128x64, 256 threads = 8 waves, wave tile 32x64 (2x4 WMMA)
// ---------------------------------------------------------------------------
enum { EPI_BIAS_F32 = 0, EPI_RESID = 1, EPI_BF16 = 2, EPI_INPROJ = 3 };

#define GBM 128
#define GBN 128
#define GBK 64
#define LDT 72  // LDS row stride in halves (144B, 16B multiple, conflict pad)

template <int EPI>
__global__ __launch_bounds__(256, 2)
void gemm_bf16(const unsigned short* __restrict__ A, const float* __restrict__ Bm,
               float* __restrict__ Cf, unsigned short* __restrict__ Cb,
               const float* __restrict__ Cin, const float* __restrict__ bias,
               const float* __restrict__ temb, int M, int N, int K) {
    __shared__ unsigned short As[GBM][LDT];  // [m][k]
    __shared__ unsigned short Bs[GBN][LDT];  // [n][k] (transposed stage)

    const int tid  = threadIdx.x;
    const int lane = tid & 31;
    const int wave = tid >> 5;
    const int wm = wave >> 1, wn = wave & 1;
    const int l15 = lane & 15, lh = lane >> 4;
    const int rowBlk = blockIdx.y * GBM;
    const int colBlk = blockIdx.x * GBN;

    v8f acc[2][4];
#pragma unroll
    for (int i = 0; i < 2; ++i)
#pragma unroll
        for (int j = 0; j < 4; ++j) acc[i][j] = vzero8();

    const int arow = tid >> 1;
    const int acol = (tid & 1) * 32;

    for (int k0 = 0; k0 < K; k0 += GBK) {
        __syncthreads();
        // --- stage A tile (bf16, contiguous b128 copies) ---
        {
            const uint4* src = (const uint4*)(A + (size_t)(rowBlk + arow) * K + k0 + acol);
            uint4* dst = (uint4*)&As[arow][acol];
#pragma unroll
            for (int j = 0; j < 4; ++j) dst[j] = src[j];
        }
        // --- stage B tile: fp32 global -> bf16, transposed to [n][k] ---
#pragma unroll
        for (int j = 0; j < 8; ++j) {
            int c  = tid + j * 256;        // 2048 float4 chunks
            int k  = c >> 5;               // 0..63
            int n0 = (c & 31) * 4;         // 0..124
            const float4 f = *(const float4*)(Bm + (size_t)(k0 + k) * N + colBlk + n0);
            Bs[n0 + 0][k] = f2bf(f.x);
            Bs[n0 + 1][k] = f2bf(f.y);
            Bs[n0 + 2][k] = f2bf(f.z);
            Bs[n0 + 3][k] = f2bf(f.w);
        }
        if (k0 + GBK < K)
            __builtin_prefetch(Bm + (size_t)(k0 + GBK) * N + colBlk, 0, 1);
        __syncthreads();

        // --- compute: 2 WMMA k-steps of 32 ---
#pragma unroll
        for (int kt = 0; kt < 2; ++kt) {
            const int kb = kt * 32 + lh * 8;
            FragBF a[2];
#pragma unroll
            for (int mi = 0; mi < 2; ++mi) {
                const unsigned short* p = &As[wm * 32 + mi * 16 + l15][kb];
                a[mi].q[0] = *(const uint4*)p;
                a[mi].q[1] = *(const uint4*)(p + 16);
            }
#pragma unroll
            for (int ni = 0; ni < 4; ++ni) {
                FragBF bf;
                const unsigned short* p = &Bs[wn * 64 + ni * 16 + l15][kb];
                bf.q[0] = *(const uint4*)p;
                bf.q[1] = *(const uint4*)(p + 16);
#pragma unroll
                for (int mi = 0; mi < 2; ++mi)
                    acc[mi][ni] = __builtin_amdgcn_wmma_f32_16x16x32_bf16(
                        false, a[mi].v, false, bf.v, (short)0, acc[mi][ni], false, false);
            }
        }
    }

    // --- epilogue ---
#pragma unroll
    for (int mi = 0; mi < 2; ++mi)
#pragma unroll
        for (int ni = 0; ni < 4; ++ni)
#pragma unroll
            for (int r = 0; r < 8; ++r) {
                int row = rowBlk + wm * 32 + mi * 16 + lh * 8 + r;
                int col = colBlk + wn * 64 + ni * 16 + l15;
                float v = acc[mi][ni][r];
                size_t idx = (size_t)row * N + col;
                if constexpr (EPI == EPI_BIAS_F32) {
                    Cf[idx] = v + bias[col];
                } else if constexpr (EPI == EPI_RESID) {
                    Cf[idx] = Cin[idx] + v;
                } else if constexpr (EPI == EPI_BF16) {
                    Cb[idx] = f2bf(v);
                } else {  // EPI_INPROJ: + bias + pos[s] + temb[b]
                    int b = row >> 10, s = row & 1023;
                    Cf[idx] = v + bias[col] + temb[(size_t)b * 1024 + col] + posemb(s, col);
                }
            }
}

// ---------------------------------------------------------------------------
// Flash attention: one block per (q-block 64, head, batch); 4 waves.
// proj pitch 11264: q @ +0, k @ +1024, v @ +2048; per-head offset h*64.
// Online softmax, all matmuls via v_wmma_f32_16x16x32_bf16.
// ---------------------------------------------------------------------------
__global__ __launch_bounds__(128)
void flash_attn(const unsigned short* __restrict__ proj, unsigned short* __restrict__ aout) {
    __shared__ unsigned short Qs[64][72];  // [m][dh]
    __shared__ unsigned short Ks[64][72];  // [kv][dh]  (K^T B-frags contiguous)
    __shared__ unsigned short Vt[64][72];  // [dh][kv]  (V B-frags contiguous)
    __shared__ unsigned short Ps[64][72];  // [m][kv]   (P A-frags)

    const int tid = threadIdx.x, lane = tid & 31, w = tid >> 5;
    const int l15 = lane & 15, lh = lane >> 4;
    const int qb = blockIdx.x, hh = blockIdx.y, b = blockIdx.z;
    const size_t pitch = 11264;
    const unsigned short* qg = proj + (size_t)(b * 1024 + qb * 64) * pitch + hh * 64;
    const unsigned short* kg = proj + (size_t)(b * 1024) * pitch + 1024 + hh * 64;
    const unsigned short* vg = kg + 1024;

    // load Q tile (64x64 bf16)
    {
        int r = tid & 63, cs = (tid >> 6) * 32;
        const uint4* src = (const uint4*)(qg + (size_t)r * pitch + cs);
        uint4* dst = (uint4*)&Qs[r][cs];
#pragma unroll
        for (int j = 0; j < 4; ++j) dst[j] = src[j];
    }

    float m_run[8], l_run[8];
    v8f o[4];
#pragma unroll
    for (int r = 0; r < 8; ++r) { m_run[r] = -__builtin_inff(); l_run[r] = 0.f; }
#pragma unroll
    for (int ni = 0; ni < 4; ++ni) o[ni] = vzero8();

    for (int kv = 0; kv < 16; ++kv) {
        __syncthreads();
        // stage K (row-major) and V (transposed)
        {
            int r = tid & 63, cs = (tid >> 6) * 32;
            const uint4* ks = (const uint4*)(kg + (size_t)(kv * 64 + r) * pitch + cs);
            uint4* kd = (uint4*)&Ks[r][cs];
#pragma unroll
            for (int j = 0; j < 4; ++j) kd[j] = ks[j];
            const uint4* vs = (const uint4*)(vg + (size_t)(kv * 64 + r) * pitch + cs);
            uint4 vv[4];
#pragma unroll
            for (int j = 0; j < 4; ++j) vv[j] = vs[j];
            const unsigned short* vh = (const unsigned short*)vv;
#pragma unroll
            for (int j = 0; j < 32; ++j) Vt[cs + j][r] = vh[j];
        }
        __syncthreads();

        // scores = Q(16x64) @ K^T(64x64) for this wave's 16-row strip
        v8f s[4];
#pragma unroll
        for (int ni = 0; ni < 4; ++ni) s[ni] = vzero8();
#pragma unroll
        for (int kt = 0; kt < 2; ++kt) {
            int kb = kt * 32 + lh * 8;
            FragBF qf;
            const unsigned short* qp = &Qs[w * 16 + l15][kb];
            qf.q[0] = *(const uint4*)qp;
            qf.q[1] = *(const uint4*)(qp + 16);
#pragma unroll
            for (int ni = 0; ni < 4; ++ni) {
                FragBF kf;
                const unsigned short* kp = &Ks[ni * 16 + l15][kb];
                kf.q[0] = *(const uint4*)kp;
                kf.q[1] = *(const uint4*)(kp + 16);
                s[ni] = __builtin_amdgcn_wmma_f32_16x16x32_bf16(
                    false, qf.v, false, kf.v, (short)0, s[ni], false, false);
            }
        }

        // online softmax (row lives in (reg r, half lh); 16-lane xor reductions)
#pragma unroll
        for (int r = 0; r < 8; ++r) {
            float x0 = s[0][r] * 0.125f, x1 = s[1][r] * 0.125f;
            float x2 = s[2][r] * 0.125f, x3 = s[3][r] * 0.125f;
            float mx = fmaxf(fmaxf(x0, x1), fmaxf(x2, x3));
#pragma unroll
            for (int off = 8; off >= 1; off >>= 1) mx = fmaxf(mx, __shfl_xor(mx, off, 32));
            float mnew  = fmaxf(m_run[r], mx);
            float alpha = __expf(m_run[r] - mnew);
            x0 = __expf(x0 - mnew); x1 = __expf(x1 - mnew);
            x2 = __expf(x2 - mnew); x3 = __expf(x3 - mnew);
            float sm = x0 + x1 + x2 + x3;
#pragma unroll
            for (int off = 8; off >= 1; off >>= 1) sm += __shfl_xor(sm, off, 32);
            l_run[r] = l_run[r] * alpha + sm;
            m_run[r] = mnew;
            o[0][r] *= alpha; o[1][r] *= alpha; o[2][r] *= alpha; o[3][r] *= alpha;
            int pr = w * 16 + lh * 8 + r;
            Ps[pr][ 0 + l15] = f2bf(x0);
            Ps[pr][16 + l15] = f2bf(x1);
            Ps[pr][32 + l15] = f2bf(x2);
            Ps[pr][48 + l15] = f2bf(x3);
        }
        __syncthreads();

        // O += P(16x64) @ V(64x64)
#pragma unroll
        for (int kt = 0; kt < 2; ++kt) {
            int kb = kt * 32 + lh * 8;
            FragBF pf;
            const unsigned short* pp = &Ps[w * 16 + l15][kb];
            pf.q[0] = *(const uint4*)pp;
            pf.q[1] = *(const uint4*)(pp + 16);
#pragma unroll
            for (int ni = 0; ni < 4; ++ni) {
                FragBF vf;
                const unsigned short* vp = &Vt[ni * 16 + l15][kb];
                vf.q[0] = *(const uint4*)vp;
                vf.q[1] = *(const uint4*)(vp + 16);
                o[ni] = __builtin_amdgcn_wmma_f32_16x16x32_bf16(
                    false, pf.v, false, vf.v, (short)0, o[ni], false, false);
            }
        }
    }

    // write merged-head output a[b, s, h*64+dh] as bf16
#pragma unroll
    for (int r = 0; r < 8; ++r) {
        float inv = 1.f / l_run[r];
        int row = qb * 64 + w * 16 + lh * 8 + r;
        size_t base = ((size_t)(b * 1024 + row)) * 1024 + hh * 64;
#pragma unroll
        for (int ni = 0; ni < 4; ++ni)
            aout[base + ni * 16 + l15] = f2bf(o[ni][r] * inv);
    }
}

// ---------------------------------------------------------------------------
// Host orchestration
// ---------------------------------------------------------------------------
extern "C" void kernel_launch(void* const* d_in, const int* in_sizes, int n_in,
                              void* d_out, int out_size, void* d_ws, size_t ws_size,
                              hipStream_t stream) {
    (void)in_sizes; (void)n_in; (void)out_size; (void)ws_size;
    const float* inputs = (const float*)d_in[0];
    const float* timev  = (const float*)d_in[1];
    const float* in_W   = (const float*)d_in[2];
    const float* in_b   = (const float*)d_in[3];
    const float* out_W  = (const float*)d_in[4];
    const float* out_b  = (const float*)d_in[5];
    const float* t_W1   = (const float*)d_in[6];
    const float* t_b1   = (const float*)d_in[7];
    const float* t_W2   = (const float*)d_in[8];
    const float* t_b2   = (const float*)d_in[9];
    const float* ln_s   = (const float*)d_in[10];
    const float* ln_b   = (const float*)d_in[11];
    const float* proj_W = (const float*)d_in[12];
    const float* attn_W = (const float*)d_in[13];
    const float* ff_W   = (const float*)d_in[14];
    float* out = (float*)d_out;

    char* ws = (char*)d_ws;
    size_t off = 0;
    auto alloc = [&](size_t bytes) -> void* {
        void* p = ws + off;
        off = (off + bytes + 255) & ~(size_t)255;
        return p;
    };
    float*          temb_hid = (float*)alloc(8ull * 4096 * 4);
    float*          temb     = (float*)alloc(8ull * 1024 * 4);
    float*          h        = (float*)alloc(8192ull * 1024 * 4);
    unsigned short* in_bf    = (unsigned short*)alloc(8192ull * 512 * 2);
    unsigned short* u_bf     = (unsigned short*)alloc(8192ull * 1024 * 2);
    unsigned short* a_bf     = (unsigned short*)alloc(8192ull * 1024 * 2);
    unsigned short* ffg_bf   = (unsigned short*)alloc(8192ull * 4096 * 2);
    unsigned short* proj_bf  = (unsigned short*)alloc(8192ull * 11264 * 2);

    // time embedding MLP
    time_mlp1<<<dim3(16, 8), 256, 0, stream>>>(timev, t_W1, t_b1, temb_hid);
    time_mlp2<<<dim3(4, 8), 256, 0, stream>>>(temb_hid, t_W2, t_b2, temb);

    // input projection + pos + temb  (h = inputs @ in_W + in_b + pos + temb)
    cvt_bf16<<<16384, 256, 0, stream>>>(inputs, in_bf, 8192 * 512);
    gemm_bf16<EPI_INPROJ><<<dim3(8, 64), 256, 0, stream>>>(
        in_bf, in_W, h, nullptr, nullptr, in_b, temb, 8192, 1024, 512);

    for (int l = 0; l < 12; ++l) {
        layernorm_bf16<<<8192, 256, 0, stream>>>(h, ln_s + l * 1024, ln_b + l * 1024, u_bf);
        // proj = u @ Wp  -> bf16 [8192, 11264]
        gemm_bf16<EPI_BF16><<<dim3(88, 64), 256, 0, stream>>>(
            u_bf, proj_W + (size_t)l * 1024 * 11264, nullptr, proj_bf, nullptr, nullptr,
            nullptr, 8192, 11264, 1024);
        // attention (flash) -> a_bf [8192, 1024]
        flash_attn<<<dim3(16, 16, 8), 128, 0, stream>>>(proj_bf, a_bf);
        // h += a @ Wa
        gemm_bf16<EPI_RESID><<<dim3(8, 64), 256, 0, stream>>>(
            a_bf, attn_W + (size_t)l * 1024 * 1024, h, nullptr, h, nullptr, nullptr,
            8192, 1024, 1024);
        // ffg = ff * gelu_tanh(gate)
        ffg_kernel<<<131072, 256, 0, stream>>>(proj_bf, ffg_bf);
        // h += ffg @ Wf
        gemm_bf16<EPI_RESID><<<dim3(8, 64), 256, 0, stream>>>(
            ffg_bf, ff_W + (size_t)l * 4096 * 1024, h, nullptr, h, nullptr, nullptr,
            8192, 1024, 4096);
    }

    // output projection: out = h @ out_W + out_b
    cvt_bf16<<<32768, 256, 0, stream>>>(h, u_bf, 8192 * 1024);
    gemm_bf16<EPI_BIAS_F32><<<dim3(2, 64), 256, 0, stream>>>(
        u_bf, out_W, out, nullptr, nullptr, out_b, nullptr, 8192, 256, 1024);
}
